// red_net_3_6493990551759
// MI455X (gfx1250) — matrix operand order repaired and checked by reading, not compiled, not run on hardware
//
#include <hip/hip_runtime.h>
#include <hip/hip_bf16.h>

typedef __attribute__((ext_vector_type(16))) _Float16 v16h;
typedef __attribute__((ext_vector_type(8)))  _Float16 v8h;
typedef __attribute__((ext_vector_type(4)))  _Float16 v4h;
typedef __attribute__((ext_vector_type(8)))  float    v8f;

#define K1   784
#define K1P  800          // K padded to multiple of 32
#define H    128
#define ROWS 32           // batch rows per block (two 16-row M tiles)
#define OUTC 10

__device__ __forceinline__ _Float16 sign_h(float v) {
    return (_Float16)((v > 0.f) ? 1.f : ((v < 0.f) ? -1.f : 0.f));
}

// ---------------- prep: binarize weights to f16 (with padding) ----------------
__global__ __launch_bounds__(256) void prep_weights(
    const float* __restrict__ W1, const float* __restrict__ W2,
    const float* __restrict__ W3,
    _Float16* __restrict__ w1h, _Float16* __restrict__ w2h,
    _Float16* __restrict__ w3h)
{
    const int i  = blockIdx.x * 256 + threadIdx.x;
    const int N1 = H * K1P;        // 102400
    const int N2 = H * H;          // 16384
    const int N3 = 16 * H;         // 2048 (rows 10..15 zero)
    if (i < N1) {
        int n = i / K1P, k = i - n * K1P;
        w1h[i] = (k < K1) ? sign_h(W1[n * K1 + k]) : (_Float16)0.f;
    } else if (i < N1 + N2) {
        int j = i - N1;
        w2h[j] = sign_h(W2[j]);
    } else if (i < N1 + N2 + N3) {
        int j = i - N1 - N2;
        int n = j / H, k = j - n * H;
        w3h[j] = (n < OUTC) ? sign_h(W3[n * H + k]) : (_Float16)0.f;
    }
}

// ---------------- fused 3-layer binarized MLP + log_softmax ----------------
__global__ __launch_bounds__(256) void fused_bnn(
    const float* __restrict__ x,
    const float* __restrict__ b1, const float* __restrict__ b2,
    const float* __restrict__ b3, const float* __restrict__ drop,
    const _Float16* __restrict__ w1h, const _Float16* __restrict__ w2h,
    const _Float16* __restrict__ w3h,
    float* __restrict__ out)
{
    __shared__ alignas(16) _Float16 xh[ROWS * K1P];   // 51200 B: x tile as f16
    __shared__ alignas(16) _Float16 a1s[ROWS * H];    //  8192 B: sign(h1)
    __shared__ alignas(16) _Float16 a2s[ROWS * H];    //  8192 B: sign(h2)
    __shared__ alignas(16) float    c3s[ROWS * 16];   //  2048 B: layer-3 logits

    const int t       = threadIdx.x;
    const int wave    = t >> 5;
    const int lane    = t & 31;
    const int m       = lane & 15;      // A row-in-tile / B,C column-in-slice
    const int hi      = lane >> 4;
    const int rowBase = blockIdx.x * ROWS;
    const int colBase = wave * 16;      // this wave's 16-col slice of the 128
    const int kbA     = hi * 8;         // A-matrix K sub-offset per lane-half
    const int kbB     = hi * 8;         // B-matrix K sub-offset per lane-half

    // ---- stage x tile (fp32 -> f16) into LDS, vectorized, zero-pad K ----
    for (int i = t; i < ROWS * (K1 / 4); i += 256) {      // 32 * 196 float4's
        int r = i / (K1 / 4), k4 = i - r * (K1 / 4);
        float4 f = *(const float4*)(x + (size_t)(rowBase + r) * K1 + 4 * k4);
        v4h h;
        h[0] = (_Float16)f.x; h[1] = (_Float16)f.y;
        h[2] = (_Float16)f.z; h[3] = (_Float16)f.w;
        *(v4h*)(&xh[r * K1P + 4 * k4]) = h;
    }
    for (int i = t; i < ROWS * 16; i += 256) {            // pad K = 784..799
        int r = i >> 4, k = K1 + (i & 15);
        xh[r * K1P + k] = (_Float16)0.f;
    }
    __syncthreads();

    // ============ layer 1: two (16x800) tiles @ shared (800x16 slice) ========
    v8f accA = {}, accB = {};
    {
        const _Float16* bcol = w1h + (size_t)(colBase + m) * K1P;
        for (int kc = 0; kc < K1P / 32; ++kc) {
            const int k0 = kc * 32;
            v8h blo = *(const v8h*)(bcol + k0 + kbB);
            v8h bhi = *(const v8h*)(bcol + k0 + kbB + 16);
            v8h a0l = *(const v8h*)(&xh[m * K1P + k0 + kbA]);
            v8h a0h = *(const v8h*)(&xh[m * K1P + k0 + 16 + kbA]);
            v8h a1l = *(const v8h*)(&xh[(16 + m) * K1P + k0 + kbA]);
            v8h a1h = *(const v8h*)(&xh[(16 + m) * K1P + k0 + 16 + kbA]);
            v16h a0, a1, b;
            #pragma unroll
            for (int i = 0; i < 8; ++i) {
                b[i]  = blo[i]; b[8 + i]  = bhi[i];
                a0[i] = a0l[i]; a0[8 + i] = a0h[i];
                a1[i] = a1l[i]; a1[8 + i] = a1h[i];
            }
            accA = __builtin_amdgcn_wmma_f32_16x16x32_f16(
                       false, a0, false, b, (short)0, accA, false, false);
            accB = __builtin_amdgcn_wmma_f32_16x16x32_f16(
                       false, a1, false, b, (short)0, accB, false, false);
        }
        const float bias = b1[colBase + m];
        #pragma unroll
        for (int v = 0; v < 8; ++v) {
            const int M = v + 8 * hi;                 // C layout: M = vgpr + 8*hi
            a1s[M * H + colBase + m]        = sign_h(accA[v] + bias);
            a1s[(16 + M) * H + colBase + m] = sign_h(accB[v] + bias);
        }
    }
    __syncthreads();

    // ============ layer 2: two (16x128) tiles @ shared (128x16 slice) ========
    v8f acc2A = {}, acc2B = {};
    {
        const _Float16* bcol = w2h + (size_t)(colBase + m) * H;
        #pragma unroll
        for (int kc = 0; kc < H / 32; ++kc) {
            const int k0 = kc * 32;
            v8h blo = *(const v8h*)(bcol + k0 + kbB);
            v8h bhi = *(const v8h*)(bcol + k0 + kbB + 16);
            v8h a0l = *(const v8h*)(&a1s[m * H + k0 + kbA]);
            v8h a0h = *(const v8h*)(&a1s[m * H + k0 + 16 + kbA]);
            v8h a1l = *(const v8h*)(&a1s[(16 + m) * H + k0 + kbA]);
            v8h a1h = *(const v8h*)(&a1s[(16 + m) * H + k0 + 16 + kbA]);
            v16h a0, a1, b;
            #pragma unroll
            for (int i = 0; i < 8; ++i) {
                b[i]  = blo[i]; b[8 + i]  = bhi[i];
                a0[i] = a0l[i]; a0[8 + i] = a0h[i];
                a1[i] = a1l[i]; a1[8 + i] = a1h[i];
            }
            acc2A = __builtin_amdgcn_wmma_f32_16x16x32_f16(
                        false, a0, false, b, (short)0, acc2A, false, false);
            acc2B = __builtin_amdgcn_wmma_f32_16x16x32_f16(
                        false, a1, false, b, (short)0, acc2B, false, false);
        }
        const float bias = b2[colBase + m];
        #pragma unroll
        for (int v = 0; v < 8; ++v) {
            const int M = v + 8 * hi;
            const float d0 = drop[(size_t)(rowBase + M) * H + colBase + m];
            const float d1 = drop[(size_t)(rowBase + 16 + M) * H + colBase + m];
            a2s[M * H + colBase + m]        = sign_h((acc2A[v] + bias) * d0);
            a2s[(16 + M) * H + colBase + m] = sign_h((acc2B[v] + bias) * d1);
        }
    }
    __syncthreads();

    // ========= layer 3 (waves 0,1): (16x128) @ (128x16pad), one M tile each ==
    if (wave < 2) {
        const int rb = 16 * wave;                     // this wave's M-tile base
        v8f acc3 = {};
        const _Float16* bcol = w3h + (size_t)m * H;   // n = m, rows 10..15 zero
        #pragma unroll
        for (int kc = 0; kc < H / 32; ++kc) {
            const int k0 = kc * 32;
            v8h alo = *(const v8h*)(&a2s[(rb + m) * H + k0 + kbA]);
            v8h ahi = *(const v8h*)(&a2s[(rb + m) * H + k0 + 16 + kbA]);
            v8h blo = *(const v8h*)(bcol + k0 + kbB);
            v8h bhi = *(const v8h*)(bcol + k0 + kbB + 16);
            v16h a, b;
            #pragma unroll
            for (int i = 0; i < 8; ++i) {
                a[i] = alo[i]; a[8 + i] = ahi[i];
                b[i] = blo[i]; b[8 + i] = bhi[i];
            }
            acc3 = __builtin_amdgcn_wmma_f32_16x16x32_f16(
                       false, a, false, b, (short)0, acc3, false, false);
        }
        const float bias = (m < OUTC) ? b3[m] : 0.f;
        #pragma unroll
        for (int v = 0; v < 8; ++v)
            c3s[(rb + v + 8 * hi) * 16 + m] = acc3[v] + bias;
    }
    __syncthreads();

    // ========= log_softmax over 10 classes, one thread per batch row =========
    if (t < ROWS) {
        const float* row = &c3s[t * 16];
        float mx = row[0];
        #pragma unroll
        for (int j = 1; j < OUTC; ++j) mx = fmaxf(mx, row[j]);
        float s = 0.f;
        #pragma unroll
        for (int j = 0; j < OUTC; ++j) s += __expf(row[j] - mx);
        const float lse = mx + __logf(s);
        #pragma unroll
        for (int j = 0; j < OUTC; ++j)
            out[(size_t)(rowBase + t) * OUTC + j] = row[j] - lse;
    }
}

extern "C" void kernel_launch(void* const* d_in, const int* in_sizes, int n_in,
                              void* d_out, int out_size, void* d_ws, size_t ws_size,
                              hipStream_t stream)
{
    const float* x    = (const float*)d_in[0];
    const float* W1   = (const float*)d_in[1];
    const float* b1   = (const float*)d_in[2];
    const float* W2   = (const float*)d_in[3];
    const float* b2   = (const float*)d_in[4];
    const float* W3   = (const float*)d_in[5];
    const float* b3   = (const float*)d_in[6];
    const float* drop = (const float*)d_in[7];
    float* out = (float*)d_out;

    char* ws = (char*)d_ws;
    _Float16* w1h = (_Float16*)(ws);                     // 128*800*2 = 204800 B
    _Float16* w2h = (_Float16*)(ws + 204800);            // 128*128*2 =  32768 B
    _Float16* w3h = (_Float16*)(ws + 204800 + 32768);    //  16*128*2 =   4096 B

    // binarize/pad weights: 102400 + 16384 + 2048 = 120832 elems = 472 * 256
    prep_weights<<<472, 256, 0, stream>>>(W1, W2, W3, w1h, w2h, w3h);

    const int B = in_sizes[0] / K1;                      // 65536
    fused_bnn<<<B / ROWS, 256, 0, stream>>>(x, b1, b2, b3, drop,
                                            w1h, w2h, w3h, out);
}
